// CausalLSTMCell_16432544874517
// MI455X (gfx1250) — matrix operand order repaired and checked by reading, not compiled, not run on hardware
//
#include <hip/hip_runtime.h>
#include <hip/hip_bf16.h>

typedef __attribute__((ext_vector_type(16))) _Float16 v16h;
typedef __attribute__((ext_vector_type(8)))  float    v8f;

#define Hdim 16
#define NNODES 6
#define NFRAGS 42   // column tiles: 6,6,6,7,8,9 per node

// ---- tree structure (compile-time) ----
constexpr int DINc[NNODES]      = {1, 1, 1, 2, 2, 2};
constexpr int IDXc[NNODES][2]   = {{0,0},{1,0},{2,0},{0,3},{1,4},{2,5}};
constexpr int NCc[NNODES]       = {0, 0, 0, 1, 2, 3};
constexpr int CHILDc[NNODES][3] = {{0,0,0},{0,0,0},{0,0,0},{0,0,0},{1,2,0},{0,3,4}};
constexpr int FRAG_BASEc[NNODES + 1] = {0, 6, 12, 18, 25, 33, 42};

// K layout of the fused GEMM: 0..15 = h_prev, 16..16+din-1 = x, 19 = bias (A has 1.0)
#define BIAS_K 19

// per node: parts 0=ifo(48 cols),1=a(16),2=n1(16),3=n2(16),4=r(nc*16)
struct NodeP {
    const float* xW[5];
    const float* xb[5];
    const float* hW[5];
    const float* hb[5];
};
struct KP {
    const float* inputs;   // (B, 6)
    const float* h;        // (6, B, 16)
    const float* c;        // (6, B, 16)
    float* out_n;          // (6, B, 16)
    float* out_h;
    float* out_c;
    int Bsz;
    int ntiles;
    NodeP np[NNODES];
};

__device__ __forceinline__ float sigf(float x) {
    // 1 / (1 + 2^(-x*log2e))
    return __builtin_amdgcn_rcpf(1.0f + __builtin_amdgcn_exp2f(-1.44269504f * x));
}
__device__ __forceinline__ float tanhfast(float x) {
    // tanh(x) = 2/(1 + e^{-2x}) - 1
    float e = __builtin_amdgcn_exp2f(-2.88539008f * x);
    return __builtin_amdgcn_rcpf(1.0f + e) * 2.0f - 1.0f;
}

__global__ __launch_bounds__(256) void tree_lstm_kernel(KP P) {
    // B fragments: [frag][lane][16 halves]; lane holds column N = lane%16;
    // lanes 0-15 carry K=0..7,16..23, lanes 16-31 carry K=8..15,24..31 (x + bias rows in lanes 0-15)
    __shared__ alignas(32) _Float16 ldsB[NFRAGS * 32 * 16];

    const int tid = threadIdx.x;

    // ---------- pack weights (+bias row) into WMMA-B layout (once per block) ----------
    for (int s = tid; s < NFRAGS * 32; s += blockDim.x) {
        const int g = s >> 5;
        const int L = s & 31;
        int i = 0;
        while (g >= FRAG_BASEc[i + 1]) ++i;
        const int tt = g - FRAG_BASEc[i];
        int part, col0 = 0;
        if (tt < 3)      { part = 0; col0 = tt * 16; }
        else if (tt < 6) { part = tt - 2; }
        else             { part = 4; col0 = (tt - 6) * 16; }
        const int cols = (part == 0) ? 48 : ((part == 4) ? NCc[i] * 16 : 16);
        const int n    = col0 + (L & 15);
        const float* Wh = P.np[i].hW[part];
        const float* Wx = P.np[i].xW[part];
        const int din   = DINc[i];
        _Float16* dst = &ldsB[(size_t)((g << 5) + L) << 4];
#pragma unroll
        for (int e = 0; e < 16; ++e) {
            const int k = (e < 8) ? (((L >> 4) << 3) + e) : (16 + ((L >> 4) << 3) + (e - 8));
            float v = 0.0f;
            if (k < 16)              v = Wh[k * cols + n];
            else if ((k - 16) < din) v = Wx[(k - 16) * cols + n];
            else if (k == BIAS_K)    v = P.np[i].hb[part][n] + P.np[i].xb[part][n];
            dst[e] = (_Float16)v;
        }
    }
    __syncthreads();

    // ---------- batch-tile loop (one wave per 16-row tile) ----------
    const int lane = tid & 31;
    const int m    = lane & 15;   // A-row / output column index
    const int half = lane >> 4;   // K-half for A, M-offset (0/8) for C/D
    const int gw   = blockIdx.x * (blockDim.x >> 5) + (tid >> 5);
    const int nw   = gridDim.x * (blockDim.x >> 5);

    for (int t = gw; t < P.ntiles; t += nw) {
        const int b0 = t * 16;
        float nk[5][8];   // n_new tiles of nodes 0..4 (children of later nodes)

#pragma unroll
        for (int i = 0; i < NNODES; ++i) {
            // --- build A: row m = [h(16) | x(din) | 0 | bias-one] as f16 16x32 ---
            const size_t hoff =
                (((size_t)i * P.Bsz + b0 + m) << 4) + ((size_t)half << 3);
            const float4 h0 = *(const float4*)(P.h + hoff);
            const float4 h1 = *(const float4*)(P.h + hoff + 4);
            float x0 = P.inputs[(size_t)(b0 + m) * 6 + IDXc[i][0]];
            float x1 = (DINc[i] > 1)
                           ? P.inputs[(size_t)(b0 + m) * 6 + IDXc[i][1]]
                           : 0.0f;
            float one = 1.0f;
            if (half) { x0 = 0.0f; x1 = 0.0f; one = 0.0f; } // K16..23 live in lanes 0-15
            v16h A = {};
            A[0]  = (_Float16)h0.x; A[1] = (_Float16)h0.y;
            A[2]  = (_Float16)h0.z; A[3] = (_Float16)h0.w;
            A[4]  = (_Float16)h1.x; A[5] = (_Float16)h1.y;
            A[6]  = (_Float16)h1.z; A[7] = (_Float16)h1.w;
            A[8]  = (_Float16)x0;   A[9] = (_Float16)x1;
            A[BIAS_K - 16 + 8] = (_Float16)one;   // element 11 <-> K=19

            auto mm = [&](int tt) -> v8f {
                const int g = FRAG_BASEc[i] + tt;
                const v16h Bf =
                    *(const v16h*)&ldsB[(size_t)((g << 5) + lane) << 4];
                const v8f zero = {};   // bias comes through the matmul; C = inline 0
                return __builtin_amdgcn_wmma_f32_16x16x32_f16(
                    false, A, false, Bf, (short)0, zero, false, false);
            };

            v8f zig = mm(0), zfg = mm(1), zog = mm(2), za = mm(3);
            v8f zn1 = mm(4), zn2 = mm(5);

            const size_t obase =
                (((size_t)i * P.Bsz + b0 + (half << 3)) << 4) + m;

            float hn[8];
#pragma unroll
            for (int v = 0; v < 8; ++v) {
                const float cp = P.c[obase + ((size_t)v << 4)];
                const float ig = sigf(zig[v]);
                const float fg = sigf(zfg[v]);
                const float og = sigf(zog[v]);
                const float a  = tanhfast(za[v]);
                const float cn = ig * a + fg * cp;
                const float hv = og * tanhfast(cn);
                hn[v] = hv;
                P.out_c[obase + ((size_t)v << 4)] = cn;
                P.out_h[obase + ((size_t)v << 4)] = hv;
            }

            float r[8];
#pragma unroll
            for (int v = 0; v < 8; ++v) r[v] = 0.0f;
#pragma unroll
            for (int j = 0; j < 3; ++j) {
                if (j < NCc[i]) {
                    v8f zr = mm(6 + j);
                    const int ch = CHILDc[i][j];
#pragma unroll
                    for (int v = 0; v < 8; ++v)
                        r[v] += sigf(zr[v]) * nk[ch][v];
                }
            }
#pragma unroll
            for (int v = 0; v < 8; ++v) {
                const float nn = sigf(zn1[v]) * r[v] + sigf(zn2[v]) * hn[v];
                if (i < 5) nk[i][v] = nn;
                P.out_n[obase + ((size_t)v << 4)] = nn;
            }
        }
    }
}

extern "C" void kernel_launch(void* const* d_in, const int* in_sizes, int n_in,
                              void* d_out, int out_size, void* d_ws, size_t ws_size,
                              hipStream_t stream) {
    (void)n_in; (void)out_size; (void)d_ws; (void)ws_size;
    static const int CHILDREN_H[NNODES] = {0, 0, 0, 1, 2, 3};

    KP P;
    P.inputs = (const float*)d_in[0];
    P.h      = (const float*)d_in[1];
    P.c      = (const float*)d_in[2];
    // d_in[3] = n : never read (all child reads refer to already-updated nodes)
    const int Bsz = in_sizes[0] / 6;
    P.Bsz    = Bsz;
    P.ntiles = Bsz / 16;
    P.out_n  = (float*)d_out;
    P.out_h  = (float*)d_out + (size_t)NNODES * Bsz * Hdim;
    P.out_c  = (float*)d_out + (size_t)2 * NNODES * Bsz * Hdim;

    // params flattened in dict-insertion order:
    // per node: ifo_x{W,b}, ifo_h{W,b}, a_x, a_h, n1_x, n1_h, n2_x, n2_h [, r_x, r_h]
    int idx = 4;
    for (int i = 0; i < NNODES; ++i) {
        NodeP& np = P.np[i];
        for (int p = 0; p < 4; ++p) {       // 0=ifo,1=a,2=n1,3=n2
            np.xW[p] = (const float*)d_in[idx++];
            np.xb[p] = (const float*)d_in[idx++];
            np.hW[p] = (const float*)d_in[idx++];
            np.hb[p] = (const float*)d_in[idx++];
        }
        if (CHILDREN_H[i] > 0) {            // 4=r
            np.xW[4] = (const float*)d_in[idx++];
            np.xb[4] = (const float*)d_in[idx++];
            np.hW[4] = (const float*)d_in[idx++];
            np.hb[4] = (const float*)d_in[idx++];
        } else {
            np.xW[4] = np.xb[4] = np.hW[4] = np.hb[4] = nullptr;
        }
    }

    dim3 grid(512), block(256);
    hipLaunchKernelGGL(tree_lstm_kernel, grid, block, 0, stream, P);
}
